// SelfAttention2d_16501264351790
// MI455X (gfx1250) — matrix operand order repaired and checked by reading, not compiled, not run on hardware
//
#include <hip/hip_runtime.h>
#include <hip/hip_bf16.h>
#include <math.h>

typedef _Float16 half_t;
typedef __attribute__((ext_vector_type(16))) _Float16 v16h;
typedef __attribute__((ext_vector_type(8)))  float    v8f;

union V16H { v16h v; _Float16 h[16]; };
union V8F  { v8f  v; float    f[8];  };

#define WMMA(a, b, c) \
  __builtin_amdgcn_wmma_f32_16x16x32_f16(false, (a), false, (b), (short)0, (c), false, false)

static constexpr int Bsz = 4, Cch = 512, Nn = 48 * 48;        // 2304
static constexpr int GROUPS = 8, HEADS = 8, Dd = Cch / HEADS; // 64
static constexpr float EPS = 1e-5f;

// ---------- WMMA fragment loaders (CDNA5 16-bit layouts, wave32) ----------
// A: 16x32 (MxK), row-major source with leading dim lda (halfs).
__device__ inline v16h load_a16(const half_t* A, int lda, int m0, int k0) {
  int lane = threadIdx.x & 31;
  int m  = m0 + (lane & 15);
  int hb = (lane >> 4) * 16;
  const half_t* p0 = A + (size_t)m * lda + k0 + hb;   // K = hb..hb+7
  const half_t* p1 = p0 + 8;                           // K = hb+8..hb+15
  V16H f;
#pragma unroll
  for (int e = 0; e < 8; ++e) { f.h[e] = p0[e]; f.h[8 + e] = p1[e]; }
  return f.v;
}

// B: 32x16 (KxN); source stored column-major as [n][k] with leading dim ldb.
__device__ inline v16h load_b16(const half_t* Bt, int ldb, int n0, int k0) {
  int lane = threadIdx.x & 31;
  int n  = n0 + (lane & 15);
  int kb = k0 + (lane >> 4) * 16;
  const half_t* p = Bt + (size_t)n * ldb + kb;
  V16H f;
#pragma unroll
  for (int e = 0; e < 16; ++e) f.h[e] = p[e];
  return f.v;
}

// ---------- helpers ----------
__global__ void k_f32tof16(const float* __restrict__ src, half_t* __restrict__ dst, int n) {
  int i = blockIdx.x * blockDim.x + threadIdx.x;
  if (i < n) dst[i] = (half_t)src[i];
}

// ---------- GroupNorm ----------
__global__ void k_gnstats(const float* __restrict__ x, float* __restrict__ stats) {
  __shared__ float s1[256], s2[256];
  int bg = blockIdx.x;                         // b*GROUPS + g
  const int total = (Cch / GROUPS) * Nn;       // 147456 contiguous floats
  const float* xp = x + (size_t)bg * total;
  float sum = 0.f, sq = 0.f;
  for (int i = threadIdx.x; i < total; i += blockDim.x) {
    float v = xp[i];
    sum += v; sq += v * v;
  }
  s1[threadIdx.x] = sum; s2[threadIdx.x] = sq;
  __syncthreads();
  for (int s = 128; s > 0; s >>= 1) {
    if ((int)threadIdx.x < s) {
      s1[threadIdx.x] += s1[threadIdx.x + s];
      s2[threadIdx.x] += s2[threadIdx.x + s];
    }
    __syncthreads();
  }
  if (threadIdx.x == 0) {
    float mu  = s1[0] / (float)total;
    float var = s2[0] / (float)total - mu * mu;
    stats[bg * 2]     = mu;
    stats[bg * 2 + 1] = rsqrtf(var + EPS);
  }
}

// h_t[b][n][c] = GN(x)  (f16, c contiguous for B-fragment loads)
__global__ void k_gnapply(const float* __restrict__ x, const float* __restrict__ gw,
                          const float* __restrict__ gb, const float* __restrict__ stats,
                          half_t* __restrict__ h_t) {
  size_t t = (size_t)blockIdx.x * blockDim.x + threadIdx.x;
  size_t total = (size_t)Bsz * Cch * Nn;
  if (t >= total) return;
  int n = (int)(t % Nn);
  int c = (int)((t / Nn) % Cch);
  int b = (int)(t / ((size_t)Nn * Cch));
  int g = c / (Cch / GROUPS);
  float mu = stats[(b * GROUPS + g) * 2];
  float rs = stats[(b * GROUPS + g) * 2 + 1];
  float v = (x[t] - mu) * rs * gw[c] + gb[c];
  h_t[((size_t)b * Nn + n) * Cch + c] = (half_t)v;
}

// ---------- QKV GEMM: 32(o) x 64(n) tile per wave ----------
__global__ void k_qkv(const half_t* __restrict__ Wq, const float* __restrict__ bias,
                      const half_t* __restrict__ h_t,
                      half_t* __restrict__ qb, half_t* __restrict__ kb, half_t* __restrict__ vb) {
  int o0 = blockIdx.x * 32, n0 = blockIdx.y * 64, b = blockIdx.z;
  const half_t* Ht = h_t + (size_t)b * Nn * Cch;
  v8f acc[2][4] = {};
  for (int kk = 0; kk < Cch; kk += 32) {
    v16h a0 = load_a16(Wq, Cch, o0,      kk);
    v16h a1 = load_a16(Wq, Cch, o0 + 16, kk);
    __builtin_prefetch(Wq + (size_t)(o0 + (threadIdx.x & 15)) * Cch + kk + 32, 0, 3);
#pragma unroll
    for (int bb = 0; bb < 4; ++bb) {
      v16h bf = load_b16(Ht, Cch, n0 + bb * 16, kk);
      acc[0][bb] = WMMA(a0, bf, acc[0][bb]);
      acc[1][bb] = WMMA(a1, bf, acc[1][bb]);
    }
    __builtin_prefetch(Ht + (size_t)(n0 + (threadIdx.x & 15)) * Cch + kk + 32, 0, 3);
  }
  int lane = threadIdx.x & 31;
  int col = lane & 15, rb = (lane >> 4) * 8;
#pragma unroll
  for (int a = 0; a < 2; ++a) {
#pragma unroll
    for (int bb = 0; bb < 4; ++bb) {
      V8F r; r.v = acc[a][bb];
#pragma unroll
      for (int i = 0; i < 8; ++i) {
        int o = o0 + a * 16 + rb + i, n = n0 + bb * 16 + col;
        float val = r.f[i] + bias[o];
        int which = o >> 9, rem = o & 511, hh = rem >> 6, dd = rem & 63;
        size_t bh = (size_t)b * HEADS + hh;
        if (which == 0)      qb[(bh * Nn + n) * Dd + dd] = (half_t)(val * 0.125f); // fold d^-0.5
        else if (which == 1) kb[(bh * Nn + n) * Dd + dd] = (half_t)val;
        else                 vb[(bh * Dd + dd) * Nn + n] = (half_t)val;
      }
    }
  }
}

// ---------- pass 1: per-row (q index j) softmax stats, 32 rows per wave ----------
__global__ void k_softmax_stats(const half_t* __restrict__ qb, const half_t* __restrict__ kb,
                                float* __restrict__ mb, float* __restrict__ zb) {
  int j0 = blockIdx.x * 32, bh = blockIdx.y;
  const half_t* Q = qb + (size_t)bh * Nn * Dd;
  const half_t* K = kb + (size_t)bh * Nn * Dd;
  v16h aq[4];
  aq[0] = load_a16(Q, Dd, j0,      0);
  aq[1] = load_a16(Q, Dd, j0,      32);
  aq[2] = load_a16(Q, Dd, j0 + 16, 0);
  aq[3] = load_a16(Q, Dd, j0 + 16, 32);
  float rmax[16], rsum[16];
#pragma unroll
  for (int r = 0; r < 16; ++r) { rmax[r] = -1e30f; rsum[r] = 0.f; }
  for (int i0 = 0; i0 < Nn; i0 += 16) {
    v16h bk0 = load_b16(K, Dd, i0, 0);
    v16h bk1 = load_b16(K, Dd, i0, 32);
#pragma unroll
    for (int t = 0; t < 2; ++t) {
      v8f s = {};
      s = WMMA(aq[t * 2],     bk0, s);
      s = WMMA(aq[t * 2 + 1], bk1, s);
      V8F sv; sv.v = s;
#pragma unroll
      for (int r = 0; r < 8; ++r) {
        float xv = sv.f[r];
        float nm = fmaxf(rmax[t * 8 + r], xv);
        rsum[t * 8 + r] = rsum[t * 8 + r] * __expf(rmax[t * 8 + r] - nm) + __expf(xv - nm);
        rmax[t * 8 + r] = nm;
      }
    }
  }
  // combine the 16 column-lanes of each half-wave
#pragma unroll
  for (int off = 1; off < 16; off <<= 1) {
#pragma unroll
    for (int r = 0; r < 16; ++r) {
      float om = __shfl_xor(rmax[r], off, 32);
      float os = __shfl_xor(rsum[r], off, 32);
      float nm = fmaxf(rmax[r], om);
      rsum[r] = rsum[r] * __expf(rmax[r] - nm) + os * __expf(om - nm);
      rmax[r] = nm;
    }
  }
  int lane = threadIdx.x & 31;
  if ((lane & 15) == 0) {
    int rb = (lane >> 4) * 8;
#pragma unroll
    for (int t = 0; t < 2; ++t)
#pragma unroll
      for (int r = 0; r < 8; ++r) {
        int j = j0 + t * 16 + rb + r;
        mb[(size_t)bh * Nn + j] = rmax[t * 8 + r];
        zb[(size_t)bh * Nn + j] = 1.0f / rsum[t * 8 + r];
      }
  }
}

// ---------- pass 2: out tile 64(d) x 32(i) per wave ----------
__global__ void k_attn_out(const half_t* __restrict__ qb, const half_t* __restrict__ kb,
                           const half_t* __restrict__ vb, const float* __restrict__ mb,
                           const float* __restrict__ zb, half_t* __restrict__ outb) {
  int i0 = blockIdx.x * 32, bh = blockIdx.y;
  int b = bh >> 3, h = bh & 7;
  const half_t* Q = qb + (size_t)bh * Nn * Dd;
  const half_t* K = kb + (size_t)bh * Nn * Dd;
  const half_t* V = vb + (size_t)bh * Dd * Nn;
  const float* mrow = mb + (size_t)bh * Nn;
  const float* zrow = zb + (size_t)bh * Nn;
  // loop-invariant K fragments for both i-tiles
  v16h bk[2][2];
  bk[0][0] = load_b16(K, Dd, i0,      0);
  bk[0][1] = load_b16(K, Dd, i0,      32);
  bk[1][0] = load_b16(K, Dd, i0 + 16, 0);
  bk[1][1] = load_b16(K, Dd, i0 + 16, 32);
  v8f acc[2][4] = {};   // [i-tile][d-tile]
  int lane = threadIdx.x & 31;
  int rb = (lane >> 4) * 8;
  bool lo = (lane < 16);
  for (int j0 = 0; j0 < Nn; j0 += 32) {
    float p[2][2][8];   // [t (j half)][i-tile][r]
#pragma unroll
    for (int t = 0; t < 2; ++t) {
      v16h aqlo = load_a16(Q, Dd, j0 + t * 16, 0);
      v16h aqhi = load_a16(Q, Dd, j0 + t * 16, 32);
#pragma unroll
      for (int it = 0; it < 2; ++it) {
        v8f s = {};
        s = WMMA(aqlo, bk[it][0], s);
        s = WMMA(aqhi, bk[it][1], s);
        V8F sv; sv.v = s;
#pragma unroll
        for (int r = 0; r < 8; ++r) {
          int j = j0 + t * 16 + rb + r;
          p[t][it][r] = __expf(sv.f[r] - mrow[j]) * zrow[j];
        }
      }
    }
    // Build P B-fragments (K = 32 rows j, 16 cols i) per i-tile. Rows 0-7 / 8-15
    // of each S tile live in opposite half-waves at the same column -> xor(16).
    V16H pf[2];
#pragma unroll
    for (int it = 0; it < 2; ++it)
#pragma unroll
      for (int r = 0; r < 8; ++r) {
        float o1 = __shfl_xor(p[0][it][r], 16, 32);
        float o2 = __shfl_xor(p[1][it][r], 16, 32);
        pf[it].h[r]     = lo ? (half_t)p[0][it][r] : (half_t)o2;
        pf[it].h[8 + r] = lo ? (half_t)o1          : (half_t)p[1][it][r];
      }
    // PV: V A-fragments amortized over both i-tiles
#pragma unroll
    for (int tt = 0; tt < 4; ++tt) {
      v16h av = load_a16(V, Nn, tt * 16, j0);
      acc[0][tt] = WMMA(av, pf[0].v, acc[0][tt]);
      acc[1][tt] = WMMA(av, pf[1].v, acc[1][tt]);
    }
  }
  int col = lane & 15;
#pragma unroll
  for (int it = 0; it < 2; ++it) {
    half_t* op = outb + ((size_t)b * Nn + (i0 + it * 16 + col)) * Cch + h * Dd;
#pragma unroll
    for (int tt = 0; tt < 4; ++tt) {
      V8F r; r.v = acc[it][tt];
#pragma unroll
      for (int e = 0; e < 8; ++e) op[tt * 16 + rb + e] = (half_t)r.f[e];
    }
  }
}

// ---------- proj GEMM + residual: 32(o) x 64(n) tile per wave ----------
__global__ void k_proj(const half_t* __restrict__ Wp, const float* __restrict__ bias,
                       const half_t* __restrict__ outb, const float* __restrict__ x,
                       float* __restrict__ y) {
  int o0 = blockIdx.x * 32, n0 = blockIdx.y * 64, b = blockIdx.z;
  const half_t* Bt = outb + (size_t)b * Nn * Cch;
  v8f acc[2][4] = {};
  for (int kk = 0; kk < Cch; kk += 32) {
    v16h a0 = load_a16(Wp, Cch, o0,      kk);
    v16h a1 = load_a16(Wp, Cch, o0 + 16, kk);
    __builtin_prefetch(Wp + (size_t)(o0 + (threadIdx.x & 15)) * Cch + kk + 32, 0, 3);
#pragma unroll
    for (int bb = 0; bb < 4; ++bb) {
      v16h bf = load_b16(Bt, Cch, n0 + bb * 16, kk);
      acc[0][bb] = WMMA(a0, bf, acc[0][bb]);
      acc[1][bb] = WMMA(a1, bf, acc[1][bb]);
    }
    __builtin_prefetch(Bt + (size_t)(n0 + (threadIdx.x & 15)) * Cch + kk + 32, 0, 3);
  }
  int lane = threadIdx.x & 31;
  int col = lane & 15, rb = (lane >> 4) * 8;
#pragma unroll
  for (int a = 0; a < 2; ++a)
#pragma unroll
    for (int bb = 0; bb < 4; ++bb) {
      V8F r; r.v = acc[a][bb];
#pragma unroll
      for (int i = 0; i < 8; ++i) {
        int o = o0 + a * 16 + rb + i, n = n0 + bb * 16 + col;
        size_t idx = ((size_t)b * Cch + o) * Nn + n;
        y[idx] = x[idx] + r.f[i] + bias[o];
      }
    }
}

extern "C" void kernel_launch(void* const* d_in, const int* in_sizes, int n_in,
                              void* d_out, int out_size, void* d_ws, size_t ws_size,
                              hipStream_t stream) {
  const float* x      = (const float*)d_in[0];
  const float* gn_w   = (const float*)d_in[1];
  const float* gn_b   = (const float*)d_in[2];
  const float* w_qkv  = (const float*)d_in[3];
  const float* b_qkv  = (const float*)d_in[4];
  const float* w_proj = (const float*)d_in[5];
  const float* b_proj = (const float*)d_in[6];
  float* out = (float*)d_out;

  // workspace carve-up (256B aligned)
  char* ws = (char*)d_ws;
  size_t off = 0;
  auto carve = [&](size_t bytes) { char* p = ws + off; off += (bytes + 255) & ~(size_t)255; return p; };
  half_t* wqkv_h  = (half_t*)carve((size_t)3 * Cch * Cch * 2);
  half_t* wproj_h = (half_t*)carve((size_t)Cch * Cch * 2);
  float*  stats   = (float*)carve((size_t)Bsz * GROUPS * 2 * 4);
  half_t* h_t     = (half_t*)carve((size_t)Bsz * Nn * Cch * 2);
  half_t* qb      = (half_t*)carve((size_t)Bsz * HEADS * Nn * Dd * 2);
  half_t* kb      = (half_t*)carve((size_t)Bsz * HEADS * Nn * Dd * 2);
  half_t* vb      = (half_t*)carve((size_t)Bsz * HEADS * Dd * Nn * 2);
  float*  mb      = (float*)carve((size_t)Bsz * HEADS * Nn * 4);
  float*  zb      = (float*)carve((size_t)Bsz * HEADS * Nn * 4);
  half_t* outb    = (half_t*)carve((size_t)Bsz * Nn * Cch * 2);
  (void)ws_size; (void)in_sizes; (void)n_in; (void)out_size;

  // weight conversion f32 -> f16
  int nwq = 3 * Cch * Cch, nwp = Cch * Cch;
  k_f32tof16<<<(nwq + 255) / 256, 256, 0, stream>>>(w_qkv, wqkv_h, nwq);
  k_f32tof16<<<(nwp + 255) / 256, 256, 0, stream>>>(w_proj, wproj_h, nwp);

  // GroupNorm
  k_gnstats<<<Bsz * GROUPS, 256, 0, stream>>>(x, stats);
  size_t total = (size_t)Bsz * Cch * Nn;
  k_gnapply<<<(unsigned)((total + 255) / 256), 256, 0, stream>>>(x, gn_w, gn_b, stats, h_t);

  // QKV projection (WMMA, 32x64 tiles)
  k_qkv<<<dim3(3 * Cch / 32, Nn / 64, Bsz), 32, 0, stream>>>(wqkv_h, b_qkv, h_t, qb, kb, vb);

  // attention (WMMA, two-pass softmax: stats then recompute+PV)
  k_softmax_stats<<<dim3(Nn / 32, Bsz * HEADS), 32, 0, stream>>>(qb, kb, mb, zb);
  k_attn_out<<<dim3(Nn / 32, Bsz * HEADS), 32, 0, stream>>>(qb, kb, vb, mb, zb, outb);

  // output projection + residual (WMMA, 32x64 tiles)
  k_proj<<<dim3(Cch / 32, Nn / 64, Bsz), 32, 0, stream>>>(wproj_h, b_proj, outb, x, out);
}
